// SVDLinearAddition_11252814316068
// MI455X (gfx1250) — matrix operand order repaired and checked by reading, not compiled, not run on hardware
//
#include <hip/hip_runtime.h>
#include <hip/hip_bf16.h>

// ---------------------------------------------------------------------------
// SVD-Linear: y = x @ ((U*(S+eps)) @ Vt)^T + bias
// bf16 WMMA GEMMs, 128x256 block tiles (8 waves, 64x64 per wave),
// double-buffered LDS filled by CDNA5 async global->LDS b128 copies.
// ---------------------------------------------------------------------------

typedef __attribute__((ext_vector_type(16))) __bf16 v16bf;
typedef __attribute__((ext_vector_type(8)))  __bf16 v8bf;
typedef __attribute__((ext_vector_type(8)))  float  v8f;
typedef int v4i __attribute__((vector_size(4 * sizeof(int))));

#define AS1 __attribute__((address_space(1)))
#define AS3 __attribute__((address_space(3)))

#define DIN   4096
#define DOUT  4096
#define RANK  4096
#define MROWS 8192   // B*T = 4*2048

#define BM   128     // block tile M
#define BN   256     // block tile N
#define BK   64      // K elements staged per buffer
#define LDA  72      // padded LDS row stride (elements); 144B = 36 banks, 16B-aligned

#if __has_builtin(__builtin_amdgcn_global_load_async_to_lds_b128)
#define USE_ASYNC 1
#else
#define USE_ASYNC 0
#endif

__device__ __forceinline__ unsigned short f2bf(float x) {
    union { float f; unsigned u; } v; v.f = x;
    unsigned r = v.u + 0x7fffu + ((v.u >> 16) & 1u);   // round-to-nearest-even
    return (unsigned short)(r >> 16);
}

// one 16-byte global -> LDS chunk (ASYNCcnt-tracked)
__device__ __forceinline__ void copy16(const unsigned short* g, unsigned short* l) {
#if USE_ASYNC
    __builtin_amdgcn_global_load_async_to_lds_b128(
        (AS1 v4i*)g, (AS3 v4i*)l, 0, 0);
#else
    *(v8bf*)l = *(const v8bf*)g;
#endif
}

__device__ __forceinline__ void wait_async_keep12() {  // retire current stage, keep prefetch
#if USE_ASYNC
#if __has_builtin(__builtin_amdgcn_s_wait_asynccnt)
    __builtin_amdgcn_s_wait_asynccnt(12);
#else
    asm volatile("s_wait_asynccnt 0xc" ::: "memory");
#endif
#endif
}
__device__ __forceinline__ void wait_async_all() {
#if USE_ASYNC
#if __has_builtin(__builtin_amdgcn_s_wait_asynccnt)
    __builtin_amdgcn_s_wait_asynccnt(0);
#else
    asm volatile("s_wait_asynccnt 0x0" ::: "memory");
#endif
#endif
}

// ---------------------------------------------------------------------------
// prep kernels
// ---------------------------------------------------------------------------
__global__ __launch_bounds__(256)
void k_prep_U(const float* __restrict__ U, const float* __restrict__ S,
              const float* __restrict__ eps, unsigned short* __restrict__ Ub) {
    size_t idx = (size_t)blockIdx.x * 256 + threadIdx.x;   // over DOUT*RANK
    int r = (int)(idx & (RANK - 1));
    Ub[idx] = f2bf(U[idx] * (S[r] + eps[r]));
}

__global__ __launch_bounds__(256)
void k_prep_VtT(const float* __restrict__ Vt, unsigned short* __restrict__ VbT) {
    __shared__ float tile[32][33];
    int bx = blockIdx.x * 32, by = blockIdx.y * 32;
    int tx = threadIdx.x, ty = threadIdx.y;          // block (32, 8)
    #pragma unroll
    for (int i = 0; i < 32; i += 8)
        tile[ty + i][tx] = Vt[(size_t)(by + ty + i) * DIN + bx + tx];
    __syncthreads();
    #pragma unroll
    for (int i = 0; i < 32; i += 8)
        VbT[(size_t)(bx + ty + i) * RANK + by + tx] = f2bf(tile[tx][ty + i]);
}

__global__ __launch_bounds__(256)
void k_prep_x(const float* __restrict__ x, unsigned short* __restrict__ xb) {
    size_t idx = (size_t)blockIdx.x * 256 + threadIdx.x;
    xb[idx] = f2bf(x[idx]);
}

// ---------------------------------------------------------------------------
// C[M][N] = A[M][K] @ B[N][K]^T  (+ bias over N when !BF16_OUT)
// 256 threads = 8 waves (2x4); block tile 128x256; wave tile 64x64 (4x4 WMMA).
// Double-buffered LDS staging via async global->LDS b128 copies.
// ---------------------------------------------------------------------------
template<bool BF16_OUT>
__global__ __launch_bounds__(256)
void k_gemm_nt(const unsigned short* __restrict__ A,
               const unsigned short* __restrict__ B,
               void* __restrict__ Cout,
               const float* __restrict__ bias,
               int M, int N, int K)
{
    __shared__ alignas(16) unsigned short As[2][BM * LDA];
    __shared__ alignas(16) unsigned short Bs[2][BN * LDA];

    const int tid   = threadIdx.x;
    const int lane  = tid & 31;
    const int wid   = tid >> 5;               // 0..7
    const int waveM = wid >> 2;               // 0..1
    const int waveN = wid & 3;                // 0..3

    const int mBase = blockIdx.x * BM + waveM * 64;
    const int nBase = blockIdx.y * BN + waveN * 64;

    const int rlo = lane & 15;                // row/col within 16
    const int hi  = lane >> 4;                // lane half select

    const unsigned short* Ablk = A + (size_t)(blockIdx.x * BM) * K;
    const unsigned short* Bblk = B + (size_t)(blockIdx.y * BN) * K;

    // per-thread copy slots (chunk c: row = c>>3, col = (c&7)*8 elements)
    // A tile: 128 rows * 8 chunks = 1024 -> 4 per thread
    // B tile: 256 rows * 8 chunks = 2048 -> 8 per thread
    auto stage_copy = [&](int buf, int kk) {
        #pragma unroll
        for (int i = 0; i < 4; ++i) {
            int c = i * 256 + tid, row = c >> 3, col = (c & 7) * 8;
            copy16(Ablk + (size_t)row * K + kk + col, &As[buf][row * LDA + col]);
        }
        #pragma unroll
        for (int i = 0; i < 8; ++i) {
            int c = i * 256 + tid, row = c >> 3, col = (c & 7) * 8;
            copy16(Bblk + (size_t)row * K + kk + col, &Bs[buf][row * LDA + col]);
        }
    };

    v8f acc[4][4];
    v8f zero = {};
    #pragma unroll
    for (int mi = 0; mi < 4; ++mi)
        #pragma unroll
        for (int nj = 0; nj < 4; ++nj) acc[mi][nj] = zero;

    stage_copy(0, 0);                         // prologue fill
    int buf = 0;

    for (int kk = 0; kk < K; kk += BK) {
        const bool has_next = (kk + BK) < K;
        if (has_next) {
            stage_copy(buf ^ 1, kk + BK);     // prefetch next stage
            wait_async_keep12();              // retire current stage's copies
        } else {
            wait_async_all();
        }
        __syncthreads();                      // LDS tiles visible to all waves

        #pragma unroll
        for (int ks = 0; ks < BK; ks += 32) {
            // A frags: lane holds row (lane&15), K chunks [kb,kb+8) & [kb+16,kb+24), kb=8*hi
            v16bf a[4];
            #pragma unroll
            for (int mi = 0; mi < 4; ++mi) {
                const unsigned short* ap =
                    &As[buf][(waveM * 64 + mi * 16 + rlo) * LDA + ks + hi * 8];
                *((v8bf*)&a[mi])     = *(const v8bf*)ap;
                *((v8bf*)&a[mi] + 1) = *(const v8bf*)(ap + 16);
            }
            // B frags: lane holds column (lane&15) = LDS row, K chunks [kb,kb+8) & [kb+16,kb+24), kb=8*hi
            v16bf b[4];
            #pragma unroll
            for (int nj = 0; nj < 4; ++nj) {
                const unsigned short* bp =
                    &Bs[buf][(waveN * 64 + nj * 16 + rlo) * LDA + ks + hi * 8];
                *((v8bf*)&b[nj])     = *(const v8bf*)bp;
                *((v8bf*)&b[nj] + 1) = *(const v8bf*)(bp + 16);
            }
            // 16 independent WMMAs per fragment group
            #pragma unroll
            for (int mi = 0; mi < 4; ++mi)
                #pragma unroll
                for (int nj = 0; nj < 4; ++nj)
                    acc[mi][nj] = __builtin_amdgcn_wmma_f32_16x16x32_bf16(
                        false, a[mi], false, b[nj],
                        (short)0, acc[mi][nj], false, false);
        }

        __syncthreads();                      // all reads done before overwrite
        buf ^= 1;
    }

    // C/D layout: VGPR i -> row (i + 8*hi), col (lane&15)
    #pragma unroll
    for (int mi = 0; mi < 4; ++mi) {
        #pragma unroll
        for (int nj = 0; nj < 4; ++nj) {
            const int col  = nBase + nj * 16 + rlo;
            const int rowb = mBase + mi * 16 + hi * 8;
            float bs = BF16_OUT ? 0.f : bias[col];
            #pragma unroll
            for (int i = 0; i < 8; ++i) {
                float val = acc[mi][nj][i] + bs;
                size_t off = (size_t)(rowb + i) * N + col;
                if (BF16_OUT) ((unsigned short*)Cout)[off] = f2bf(val);
                else          ((float*)Cout)[off]          = val;
            }
        }
    }
}

extern "C" void kernel_launch(void* const* d_in, const int* in_sizes, int n_in,
                              void* d_out, int out_size, void* d_ws, size_t ws_size,
                              hipStream_t stream) {
    const float* x    = (const float*)d_in[0];   // [4,2048,4096]
    const float* U    = (const float*)d_in[1];   // [4096,4096]
    const float* S    = (const float*)d_in[2];   // [4096]
    const float* Vt   = (const float*)d_in[3];   // [4096,4096]
    const float* eps  = (const float*)d_in[4];   // [4096]
    const float* bias = (const float*)d_in[5];   // [4096]
    float* y = (float*)d_out;                    // [8192,4096] f32

    // workspace layout (bytes)
    char* ws = (char*)d_ws;
    unsigned short* Ub  = (unsigned short*)(ws);                           // 32 MB
    unsigned short* VbT = (unsigned short*)(ws + (((size_t)32) << 20));    // 32 MB
    unsigned short* Wb  = (unsigned short*)(ws + (((size_t)64) << 20));    // 32 MB
    unsigned short* xb  = (unsigned short*)(ws + (((size_t)96) << 20));    // 64 MB

    // 1) precision/layout prep
    k_prep_U  <<<(DOUT * (size_t)RANK) / 256, 256, 0, stream>>>(U, S, eps, Ub);
    k_prep_VtT<<<dim3(DIN / 32, RANK / 32), dim3(32, 8), 0, stream>>>(Vt, VbT);
    k_prep_x  <<<((size_t)MROWS * DIN) / 256, 256, 0, stream>>>(x, xb);

    // 2) w[o][c] = sum_r Ub[o][r] * VbT[c][r]   (bf16 out)
    k_gemm_nt<true><<<dim3(DOUT / BM, DIN / BN), 256, 0, stream>>>(
        Ub, VbT, (void*)Wb, nullptr, DOUT, DIN, RANK);

    // 3) y[m][o] = sum_k xb[m][k] * Wb[o][k] + bias[o]   (f32 out)
    k_gemm_nt<false><<<dim3(MROWS / BM, DOUT / BN), 256, 0, stream>>>(
        xb, Wb, (void*)y, bias, MROWS, DOUT, DIN);
}
